// PFNet_68401649156591
// MI455X (gfx1250) — matrix operand order repaired and checked by reading, not compiled, not run on hardware
//
#include <hip/hip_runtime.h>
#include <math.h>

typedef __attribute__((ext_vector_type(16))) _Float16 v16h;
typedef __attribute__((ext_vector_type(8)))  _Float16 v8h;
typedef __attribute__((ext_vector_type(8)))  float    v8f;
typedef _Float16 h16;

#define B_     4
#define N_     12800
#define P_     14
#define F_     22
#define FP_    32        // F padded to WMMA K multiple
#define DD_    128
#define NROT_  100
#define NBTOT_ 200       // 2*NROT_
#define NBINS_ 20
#define BS_    640
#define KNN_   16
#define E_     (NBINS_*BS_*KNN_)   // 204800 edges per batch
#define H_     256
#define O_     32
#define EF_    45
#define EFP_   64        // edge feature dim padded
#define BN_    (B_*N_)   // 51200
#define BE_    (B_*E_)   // 819200

enum { EP_NONE = 0, EP_BIAS = 1, EP_BIAS_ELU = 2, EP_BIAS_SIGMOID = 3, EP_ROWNORM = 4 };

// ---------------- WMMA helpers (CDNA5 16x16x32 f16 layouts) ----------------

__device__ __forceinline__ v8f wmma16(v16h a, v16h b, v8f c) {
  return __builtin_amdgcn_wmma_f32_16x16x32_f16(false, a, false, b, (short)0, c,
                                                false, false);
}

__device__ __forceinline__ v16h cat8(v8h lo, v8h hi) {
  return __builtin_shufflevector(lo, hi, 0, 1, 2, 3, 4, 5, 6, 7,
                                         8, 9, 10, 11, 12, 13, 14, 15);
}

// A fragment (16x32): lane l = row M, k = h + hi*16 contiguous.
// base points at row 0 of the 16-row strip; one contiguous 32B run per lane.
__device__ __forceinline__ v16h frag_a(const h16* base, int ld, int k0) {
  int l = threadIdx.x & 15, hi = (threadIdx.x >> 4) & 1;
  const h16* p = base + (size_t)l * ld + k0 + hi * 16;
  v8h lo = *(const v8h*)p;
  v8h hb = *(const v8h*)(p + 8);
  return cat8(lo, hb);
}

// B fragment from weights pre-swizzled to [K/32][N/16][lane][16 halfs].
__device__ __forceinline__ v16h frag_b_sw(const h16* Bf, int kt, int nt, int ntn) {
  int lane = threadIdx.x & 31;
  const h16* p = Bf + (((size_t)kt * ntn + nt) * 32 + lane) * 16;
  v8h lo = *(const v8h*)p;
  v8h hb = *(const v8h*)(p + 8);
  return cat8(lo, hb);
}

// B = M^T fragment from a row-major f16 matrix (rows become columns):
// lane l = output col -> row (col0+l) of M; k = h + hi*16 contiguous.
__device__ __forceinline__ v16h frag_bT(const h16* Mrow0, int ld, int k0) {
  int l = threadIdx.x & 15, hi = (threadIdx.x >> 4) & 1;
  const h16* p = Mrow0 + (size_t)l * ld + k0 + hi * 16;
  v8h lo = *(const v8h*)p;
  v8h hb = *(const v8h*)(p + 8);
  return cat8(lo, hb);
}

// ------- generic WMMA GEMM, fully unrolled & guard-free: C = A @ B ---------
// NT: 16-col tiles per block (all valid by launch geometry), KT: K/32.
// Block = 128 threads = 4 waves, each wave owns a 16-row strip of 64 rows.

template <int NT, int KT>
__global__ __launch_bounds__(128) void gemm_wmma(
    const h16* __restrict__ A, const h16* __restrict__ Bf,
    const float* __restrict__ bias, const float* __restrict__ deg,
    float* __restrict__ C, h16* __restrict__ Ch,
    int M, int N, int epil)
{
  constexpr int K = KT * 32;
  int wv = threadIdx.x >> 5;
  int rowBase = blockIdx.y * 64 + wv * 16;
  int colBase = blockIdx.x * (NT * 16);
  int ntn = N >> 4;
  int nt0 = colBase >> 4;
  v8f acc[NT] = {};

  const h16* Arow = A + (size_t)rowBase * K;
#pragma unroll
  for (int kt = 0; kt < KT; ++kt) {
    v16h af = frag_a(Arow, K, kt * 32);
#pragma unroll
    for (int t = 0; t < NT; ++t) {
      v16h bf = frag_b_sw(Bf, kt, nt0 + t, ntn);
      acc[t] = wmma16(af, bf, acc[t]);
    }
  }

  int l = threadIdx.x & 15, hi = (threadIdx.x >> 4) & 1;
#pragma unroll
  for (int t = 0; t < NT; ++t) {
    int col = colBase + t * 16 + l;
#pragma unroll
    for (int r = 0; r < 8; ++r) {
      int row = rowBase + r + hi * 8;
      float v = acc[t][r];
      if (epil == EP_BIAS)              { v += bias[col]; }
      else if (epil == EP_BIAS_ELU)     { v += bias[col]; v = v > 0.f ? v : expm1f(v); }
      else if (epil == EP_BIAS_SIGMOID) { v += bias[col]; v = 1.f / (1.f + __expf(-v)); }
      else if (epil == EP_ROWNORM)      { v *= rsqrtf(deg[row] + 1e-6f); }
      if (C)  C [(size_t)row * N + col] = v;
      if (Ch) Ch[(size_t)row * N + col] = (h16)v;
    }
  }
}

// ---------------- small utility kernels ----------------

__global__ void fill_zero_f32(float* p, int n) {
  int i = blockIdx.x * blockDim.x + threadIdx.x;
  if (i < n) p[i] = 0.f;
}
__global__ void fill_zero_i32(int* p, int n) {
  int i = blockIdx.x * blockDim.x + threadIdx.x;
  if (i < n) p[i] = 0;
}

// f32 [K,N] -> f16 fragment-order [Kp/32][N/16][lane 0..31][16 halfs],
// zero-padding K to Kp. lane=(hi*16+l): col n = nt*16+l, k = kt*32 + h + hi*16.
__global__ void cvt_swz(const float* __restrict__ src, h16* __restrict__ dst,
                        int K, int N, int Kp) {
  int i = blockIdx.x * blockDim.x + threadIdx.x;
  int total = (Kp >> 5) * (N >> 4) * 32 * 16;
  if (i >= total) return;
  int h    = i & 15;
  int lane = (i >> 4) & 31;
  int tile = i >> 9;
  int ntn  = N >> 4;
  int kt = tile / ntn, nt = tile - kt * ntn;
  int l = lane & 15, hi = lane >> 4;
  int k = kt * 32 + h + hi * 16;
  int n = nt * 16 + l;
  dst[i] = (k < K) ? (h16)src[(size_t)k * N + n] : (h16)0.f;
}

__global__ void build_xenc(const float* __restrict__ Xprop, const int* __restrict__ Xid,
                           float* __restrict__ xe32, h16* __restrict__ xe16) {
  int i = blockIdx.x * blockDim.x + threadIdx.x;
  if (i >= BN_) return;
  int id = Xid[i];
  float* o32 = xe32 + (size_t)i * F_;
  h16*   o16 = xe16 + (size_t)i * FP_;
#pragma unroll
  for (int c = 0; c < 8; ++c) { float v = (c == id) ? 1.f : 0.f; o32[c] = v; o16[c] = (h16)v; }
  const float* xp = Xprop + (size_t)i * P_;
#pragma unroll
  for (int c = 0; c < P_; ++c) { float v = xp[c]; o32[8 + c] = v; o16[8 + c] = (h16)v; }
#pragma unroll
  for (int c = F_; c < FP_; ++c) o16[c] = (h16)0.f;
}

// c = pe @ rot ; bin = argmax([c, -c]) -- one wave per node
__global__ __launch_bounds__(128) void bin_assign(const float* __restrict__ pe,
                                                  const float* __restrict__ rot,
                                                  int* __restrict__ binidx) {
  int wvg  = (blockIdx.x * blockDim.x + threadIdx.x) >> 5;
  int lane = threadIdx.x & 31;
  if (wvg >= BN_) return;
  const float* pr = pe + (size_t)wvg * DD_;
  float bestv = -INFINITY; int besti = 0x7fffffff;
  for (int j = lane; j < NROT_; j += 32) {
    float c = 0.f;
    for (int k = 0; k < DD_; ++k) c += pr[k] * rot[k * NROT_ + j];
    if (c > bestv || (c == bestv && j < besti)) { bestv = c; besti = j; }
    float m = -c; int mi = j + NROT_;
    if (m > bestv || (m == bestv && mi < besti)) { bestv = m; besti = mi; }
  }
  for (int off = 16; off > 0; off >>= 1) {
    float ov = __shfl_xor(bestv, off, 32);
    int   oi = __shfl_xor(besti, off, 32);
    if (ov > bestv || (ov == bestv && oi < besti)) { bestv = ov; besti = oi; }
  }
  if (lane == 0) binidx[wvg] = besti;
}

__global__ void bin_count(const int* __restrict__ binidx, int* __restrict__ counts) {
  int i = blockIdx.x * blockDim.x + threadIdx.x;
  if (i >= BN_) return;
  int b = i / N_;
  atomicAdd(&counts[b * NBTOT_ + binidx[i]], 1);
}

__global__ void bin_scan(const int* __restrict__ counts, int* __restrict__ offs) {
  int b = threadIdx.x;
  if (b >= B_) return;
  int run = 0;
  for (int j = 0; j < NBTOT_; ++j) { offs[b * NBTOT_ + j] = run; run += counts[b * NBTOT_ + j]; }
}

// stable compaction: one wave per (batch, bin); ballot-prefix == stable argsort
__global__ __launch_bounds__(128) void bin_scatter(const int* __restrict__ binidx,
                                                   const int* __restrict__ offs,
                                                   int* __restrict__ split) {
  int wvg  = (blockIdx.x * blockDim.x + threadIdx.x) >> 5;
  int lane = threadIdx.x & 31;
  if (wvg >= B_ * NBTOT_) return;
  int b = wvg / NBTOT_, bin = wvg % NBTOT_;
  int base = offs[wvg];
  for (int n0 = 0; n0 < N_; n0 += 32) {
    int n = n0 + lane;
    bool pred = (binidx[b * N_ + n] == bin);
    unsigned mask = (unsigned)__ballot(pred);
    if (pred) {
      int pos = base + __popc(mask & ((1u << lane) - 1u));
      split[b * N_ + pos] = n;
    }
    base += __popc(mask);
  }
}

// gather pe rows into bin order (f16) + squared norms -- one wave per slot
__global__ __launch_bounds__(128) void gather_sub(const h16* __restrict__ pe16,
                                                  const float* __restrict__ pe32,
                                                  const int* __restrict__ split,
                                                  h16* __restrict__ sub, float* __restrict__ na) {
  int slot = (blockIdx.x * blockDim.x + threadIdx.x) >> 5;
  int lane = threadIdx.x & 31;
  if (slot >= BN_) return;
  int b = slot / N_;
  int node = split[slot];
  const h16*   sp = pe16 + ((size_t)b * N_ + node) * DD_;
  const float* fp = pe32 + ((size_t)b * N_ + node) * DD_;
  h16* dp = sub + (size_t)slot * DD_;
  float s = 0.f;
  for (int j = lane; j < DD_; j += 32) { dp[j] = sp[j]; float v = fp[j]; s += v * v; }
  for (int off = 16; off > 0; off >>= 1) s += __shfl_xor(s, off, 32);
  if (lane == 0) na[slot] = s;
}

// per block: 16 rows of one bin. WMMA Gram (16x640, K=128) -> exp-dist strip in
// LDS -> iterative top-16 per row (ties -> lower index, matching lax.top_k).
__global__ __launch_bounds__(128) void gram_topk(const h16* __restrict__ sub,
                                                 const float* __restrict__ na,
                                                 const int* __restrict__ split,
                                                 int* __restrict__ srcA, int* __restrict__ dstA,
                                                 float* __restrict__ dval) {
  __shared__ float dm[16 * BS_];   // 40 KB
  __shared__ float nrow[16];
  int tid = threadIdx.x, wv = tid >> 5, lane = tid & 31;
  int l = tid & 15, hi = (tid >> 4) & 1;
  int blk = blockIdx.x;            // B*NBINS*(BS/16)
  int rt  = blk % (BS_ / 16);
  int bb  = blk / (BS_ / 16);      // b*NBINS + bin
  int b   = bb / NBINS_, bin = bb % NBINS_;
  int binSlot = bb * BS_;
  int row0 = rt * 16;

  if (tid < 16) nrow[tid] = na[binSlot + row0 + tid];
  __syncthreads();

  const h16* Arow = sub + (size_t)(binSlot + row0) * DD_;
  v16h afr[4];
#pragma unroll
  for (int ks = 0; ks < 4; ++ks) afr[ks] = frag_a(Arow, DD_, ks * 32);

  for (int ct = wv; ct < BS_ / 16; ct += 4) {
    int col0 = ct * 16;
    const h16* Brow = sub + (size_t)(binSlot + col0) * DD_;
    v8f acc = {};
#pragma unroll
    for (int ks = 0; ks < 4; ++ks) {
      v16h bf = frag_bT(Brow, DD_, ks * 32);   // B = sub^T, contiguous per lane
      acc = wmma16(afr[ks], bf, acc);
    }
    int col = col0 + l;
    float nc = na[binSlot + col];
#pragma unroll
    for (int r = 0; r < 8; ++r) {
      int rl = r + hi * 8;
      float d2 = fmaxf(nrow[rl] - 2.f * acc[r] + nc, 1e-6f);
      dm[rl * BS_ + col] = __expf(-0.1f * sqrtf(d2));
    }
  }
  __syncthreads();

  for (int rr = 0; rr < 4; ++rr) {
    int rl = wv + rr * 4;
    float* drow = dm + rl * BS_;
    for (int it = 0; it < KNN_; ++it) {
      float bv = -INFINITY; int bi = 0x7fffffff;
      for (int cc = lane; cc < BS_; cc += 32) {
        float v = drow[cc];
        if (v > bv || (v == bv && cc < bi)) { bv = v; bi = cc; }
      }
      for (int off = 16; off > 0; off >>= 1) {
        float ov = __shfl_xor(bv, off, 32);
        int   oi = __shfl_xor(bi, off, 32);
        if (ov > bv || (ov == bv && oi < bi)) { bv = ov; bi = oi; }
      }
      if (lane == 0) {
        size_t e = (size_t)b * E_ + (size_t)(bin * BS_ + row0 + rl) * KNN_ + it;
        dval[e] = bv;
        srcA[e] = split[b * N_ + bin * BS_ + row0 + rl];
        dstA[e] = split[b * N_ + bin * BS_ + bi];
        drow[bi] = -INFINITY;
      }
      __syncthreads();   // uniform trip counts across all 4 waves
    }
  }
}

// fused edge FFN: 16 edges per block; layer1 WMMA (K=64, N=128) + elu,
// layer2 (128->1) fused as a dot with edge_W2 in the epilogue, sigmoid,
// |ev| scattered into deg.
__global__ __launch_bounds__(128) void edge_ffn(
    const float* __restrict__ xenc32, const int* __restrict__ srcA,
    const int* __restrict__ dstA, const float* __restrict__ dval,
    const h16* __restrict__ W1f, const float* __restrict__ b1,
    const float* __restrict__ W2, const float* __restrict__ b2v,
    float* __restrict__ ev, float* __restrict__ deg)
{
  __shared__ h16 sA[16 * EFP_];
  __shared__ float rowsum[16];
  int tid = threadIdx.x, wv = tid >> 5, lane = tid & 31;
  int hi = (tid >> 4) & 1;
  size_t e0 = (size_t)blockIdx.x * 16;
  int b = (int)(e0 / E_);

  if (tid < 16) rowsum[tid] = 0.f;
  {
    int er = tid >> 3;
    int j0 = (tid & 7) * 8;
    size_t e = e0 + er;
    int sn = srcA[e], dn = dstA[e];
    const float* xs = xenc32 + ((size_t)b * N_ + sn) * F_;
    const float* xd = xenc32 + ((size_t)b * N_ + dn) * F_;
    float dv = dval[e];
#pragma unroll
    for (int j = 0; j < 8; ++j) {
      int jj = j0 + j;
      float v = 0.f;
      if (jj < F_)            v = xs[jj];
      else if (jj < 2 * F_)   v = xd[jj - F_];
      else if (jj == 2 * F_)  v = dv;
      sA[er * EFP_ + jj] = (h16)v;
    }
  }
  __syncthreads();

  v16h af[2];
#pragma unroll
  for (int ks = 0; ks < 2; ++ks) af[ks] = frag_a(sA, EFP_, ks * 32);

  float part[8];
#pragma unroll
  for (int r = 0; r < 8; ++r) part[r] = 0.f;

#pragma unroll
  for (int tt = 0; tt < 2; ++tt) {
    int nt = wv * 2 + tt;
    v8f acc = {};
#pragma unroll
    for (int ks = 0; ks < 2; ++ks) {
      v16h bf = frag_b_sw(W1f, ks, nt, 128 / 16);
      acc = wmma16(af[ks], bf, acc);
    }
    int col = nt * 16 + (tid & 15);
    float w2 = W2[col];
    float bb = b1[col];
#pragma unroll
    for (int r = 0; r < 8; ++r) {
      float v = acc[r] + bb;
      v = v > 0.f ? v : expm1f(v);
      part[r] += v * w2;
    }
  }
#pragma unroll
  for (int off = 1; off < 16; off <<= 1)
#pragma unroll
    for (int r = 0; r < 8; ++r) part[r] += __shfl_xor(part[r], off, 32);
  if ((lane & 15) == 0)
#pragma unroll
    for (int r = 0; r < 8; ++r) atomicAdd(&rowsum[r + hi * 8], part[r]);
  __syncthreads();
  if (tid < 16) {
    float s = rowsum[tid] + b2v[0];
    float sg = 1.f / (1.f + __expf(-s));
    size_t e = e0 + tid;
    ev[e] = sg;
    atomicAdd(&deg[(size_t)b * N_ + srcA[e]], fabsf(sg));
  }
}

// agg[src] += fhom_scaled[dst] * ev -- one wave per edge
__global__ __launch_bounds__(128) void msg_scatter(const float* __restrict__ fhom,
                                                   const int* __restrict__ srcA,
                                                   const int* __restrict__ dstA,
                                                   const float* __restrict__ ev,
                                                   float* __restrict__ agg)
{
  int eg = (blockIdx.x * blockDim.x + threadIdx.x) >> 5;
  int lane = threadIdx.x & 31;
  if (eg >= BE_) return;
  size_t e = (size_t)eg;
  int b = eg / E_;
  float w = ev[e];
  const float* srow = fhom + ((size_t)b * N_ + dstA[e]) * H_;
  float* drow = agg + ((size_t)b * N_ + srcA[e]) * H_;
  for (int h = lane; h < H_; h += 32) atomicAdd(&drow[h], srow[h] * w);
}

// g = elu(gate * agg*norm + (1-gate) * f_het)
__global__ void combine(const float* __restrict__ agg, const float* __restrict__ deg,
                        const float* __restrict__ t1, const float* __restrict__ t2,
                        h16* __restrict__ g16)
{
  long long i = (long long)blockIdx.x * blockDim.x + threadIdx.x;
  if (i >= (long long)BN_ * H_) return;
  long long row = i / H_;
  float norm = rsqrtf(deg[row] + 1e-6f);
  float gate = t2[i];
  float v = gate * (agg[i] * norm) + (1.f - gate) * t1[i];
  v = v > 0.f ? v : expm1f(v);
  g16[i] = (h16)v;
}

// ---------------- launcher ----------------

static inline size_t align_up(size_t x) { return (x + 255) & ~(size_t)255; }

extern "C" void kernel_launch(void* const* d_in, const int* in_sizes, int n_in,
                              void* d_out, int out_size, void* d_ws, size_t ws_size,
                              hipStream_t stream)
{
  (void)in_sizes; (void)n_in; (void)out_size; (void)ws_size;
  const float* Xprop = (const float*)d_in[0];
  const int*   Xid   = (const int*)d_in[1];
  const float* encW1 = (const float*)d_in[2];
  const float* encB1 = (const float*)d_in[3];
  const float* encW2 = (const float*)d_in[4];
  const float* encB2 = (const float*)d_in[5];
  const float* rot   = (const float*)d_in[6];
  const float* edgW1 = (const float*)d_in[7];
  const float* edgB1 = (const float*)d_in[8];
  const float* edgW2 = (const float*)d_in[9];
  const float* edgB2 = (const float*)d_in[10];
  const float* nodW  = (const float*)d_in[11];
  const float* nodB  = (const float*)d_in[12];
  const float* theta = (const float*)d_in[13];
  const float* Wh    = (const float*)d_in[14];
  const float* Wt    = (const float*)d_in[15];
  const float* bt    = (const float*)d_in[16];
  const float* decW  = (const float*)d_in[17];
  const float* decB  = (const float*)d_in[18];
  float* out = (float*)d_out;

  char* ws = (char*)d_ws;
  size_t off = 0;
  auto alloc = [&](size_t bytes) -> char* {
    char* p = ws + off;
    off = align_up(off + bytes);
    return p;
  };

  float* xenc32 = (float*)alloc((size_t)BN_ * F_ * 4);
  h16*   xenc16 = (h16*)  alloc((size_t)BN_ * FP_ * 2);
  h16*   hbuf   = (h16*)  alloc((size_t)BN_ * 128 * 2);
  float* pe32   = (float*)alloc((size_t)BN_ * DD_ * 4);
  h16*   pe16   = (h16*)  alloc((size_t)BN_ * DD_ * 2);
  int*   binidx = (int*)  alloc((size_t)BN_ * 4);
  int*   counts = (int*)  alloc((size_t)B_ * NBTOT_ * 4);
  int*   offs   = (int*)  alloc((size_t)B_ * NBTOT_ * 4);
  int*   split  = (int*)  alloc((size_t)BN_ * 4);
  h16*   sub16  = (h16*)  alloc((size_t)BN_ * DD_ * 2);
  float* na     = (float*)alloc((size_t)BN_ * 4);
  int*   srcA   = (int*)  alloc((size_t)BE_ * 4);
  int*   dstA   = (int*)  alloc((size_t)BE_ * 4);
  float* dval   = (float*)alloc((size_t)BE_ * 4);
  float* ev     = (float*)alloc((size_t)BE_ * 4);
  float* deg    = (float*)alloc((size_t)BN_ * 4);
  h16*   x16    = (h16*)  alloc((size_t)BN_ * H_ * 2);
  float* fhom   = (float*)alloc((size_t)BN_ * H_ * 4);
  float* agg    = (float*)alloc((size_t)BN_ * H_ * 4);
  float* t1     = (float*)alloc((size_t)BN_ * H_ * 4);
  float* t2     = (float*)alloc((size_t)BN_ * H_ * 4);
  h16*   g16b   = (h16*)  alloc((size_t)BN_ * H_ * 2);
  h16* encW1f = (h16*)alloc((size_t)FP_  * 128 * 2);
  h16* encW2f = (h16*)alloc((size_t)128  * 128 * 2);
  h16* nodWf  = (h16*)alloc((size_t)FP_  * H_  * 2);
  h16* thetaf = (h16*)alloc((size_t)H_   * H_  * 2);
  h16* Whf    = (h16*)alloc((size_t)H_   * H_  * 2);
  h16* Wtf    = (h16*)alloc((size_t)H_   * H_  * 2);
  h16* edgW1f = (h16*)alloc((size_t)EFP_ * 128 * 2);
  h16* decWf  = (h16*)alloc((size_t)H_   * O_  * 2);

  // zero accumulators
  fill_zero_i32<<<(B_ * NBTOT_ + 255) / 256, 256, 0, stream>>>(counts, B_ * NBTOT_);
  fill_zero_f32<<<(BN_ + 255) / 256, 256, 0, stream>>>(deg, BN_);
  fill_zero_f32<<<(BN_ * H_ + 255) / 256, 256, 0, stream>>>(agg, BN_ * H_);

  auto cvt = [&](const float* s, h16* d, int K, int N, int Kp) {
    int n = (Kp / 32) * (N / 16) * 32 * 16;
    cvt_swz<<<(n + 255) / 256, 256, 0, stream>>>(s, d, K, N, Kp);
  };
  cvt(encW1, encW1f, F_, 128, FP_);
  cvt(encW2, encW2f, 128, 128, 128);
  cvt(nodW,  nodWf,  F_, H_,  FP_);
  cvt(theta, thetaf, H_, H_,  H_);
  cvt(Wh,    Whf,    H_, H_,  H_);
  cvt(Wt,    Wtf,    H_, H_,  H_);
  cvt(edgW1, edgW1f, EF_, 128, EFP_);
  cvt(decW,  decWf,  H_, O_,  H_);

  build_xenc<<<(BN_ + 255) / 256, 256, 0, stream>>>(Xprop, Xid, xenc32, xenc16);

  // guard-free GEMM dispatch: pick <NT,KT> so every block is full.
  auto gemm = [&](const h16* A, const h16* Bf, const float* bias, const float* degp,
                  float* C, h16* Ch, int M, int N, int K, int epil) {
    if (N % 64 == 0) {
      dim3 grid((unsigned)(N / 64), (unsigned)(M / 64));
      if (K == 32)
        gemm_wmma<4, 1><<<grid, 128, 0, stream>>>(A, Bf, bias, degp, C, Ch, M, N, epil);
      else if (K == 128)
        gemm_wmma<4, 4><<<grid, 128, 0, stream>>>(A, Bf, bias, degp, C, Ch, M, N, epil);
      else
        gemm_wmma<4, 8><<<grid, 128, 0, stream>>>(A, Bf, bias, degp, C, Ch, M, N, epil);
    } else {            // N == 32 decoder
      dim3 grid((unsigned)(N / 32), (unsigned)(M / 64));
      gemm_wmma<2, 8><<<grid, 128, 0, stream>>>(A, Bf, bias, degp, C, Ch, M, N, epil);
    }
  };

  // encoder FFN -> pe
  gemm(xenc16, encW1f, encB1, nullptr, nullptr, hbuf, BN_, 128, FP_, EP_BIAS_ELU);
  gemm(hbuf,   encW2f, encB2, nullptr, pe32, pe16, BN_, DD_, 128, EP_BIAS);

  // LSH binning + stable counting sort
  bin_assign <<<BN_ / 4, 128, 0, stream>>>(pe32, rot, binidx);
  bin_count  <<<(BN_ + 255) / 256, 256, 0, stream>>>(binidx, counts);
  bin_scan   <<<1, 64, 0, stream>>>(counts, offs);
  bin_scatter<<<(B_ * NBTOT_) / 4, 128, 0, stream>>>(binidx, offs, split);

  // per-bin kNN graph
  gather_sub<<<BN_ / 4, 128, 0, stream>>>(pe16, pe32, split, sub16, na);
  gram_topk <<<B_ * NBINS_ * (BS_ / 16), 128, 0, stream>>>(sub16, na, split,
                                                           srcA, dstA, dval);
  // edge weights (+deg)
  edge_ffn<<<BE_ / 16, 128, 0, stream>>>(xenc32, srcA, dstA, dval,
                                         edgW1f, edgB1, edgW2, edgB2, ev, deg);

  // node encoder + gated graph conv + decoder
  gemm(xenc16, nodWf, nodB, nullptr, nullptr, x16, BN_, H_, FP_, EP_BIAS_ELU);
  gemm(x16, thetaf, nullptr, deg, fhom, nullptr, BN_, H_, H_, EP_ROWNORM);
  msg_scatter<<<BE_ / 4, 128, 0, stream>>>(fhom, srcA, dstA, ev, agg);
  gemm(x16, Whf, nullptr, nullptr, t1, nullptr, BN_, H_, H_, EP_NONE);
  gemm(x16, Wtf, bt,      nullptr, t2, nullptr, BN_, H_, H_, EP_BIAS_SIGMOID);
  combine<<<(BN_ * H_ + 255) / 256, 256, 0, stream>>>(agg, deg, t1, t2, g16b);
  gemm(g16b, decWf, decB, nullptr, out, nullptr, BN_, O_, H_, EP_BIAS);
}
